// WanAttention_386547057223
// MI455X (gfx1250) — matrix operand order repaired and checked by reading, not compiled, not run on hardware
//
#include <hip/hip_runtime.h>
#include <math.h>

// ---------------------------------------------------------------------------
// Problem constants (from the reference)
// ---------------------------------------------------------------------------
#define S_TOK  3072
#define DIMW   1536
#define HEADS  12
#define DHEAD  128
#define EPSV   1e-5f

typedef __bf16 bf16_t;
typedef __bf16 v16bf __attribute__((ext_vector_type(16)));
typedef __bf16 v8bf  __attribute__((ext_vector_type(8)));
typedef float  v8f   __attribute__((ext_vector_type(8)));

// Tensor Data Mover availability (probe-verified builtin; arity differs by
// toolchain: therock headers ship <hip/amd_detail/amd_gfx1250_TDM.h> -> 6 args)
#if defined(__gfx1250__) && __has_builtin(__builtin_amdgcn_tensor_load_to_lds)
#define USE_TDM 1
#else
#define USE_TDM 0
#endif
#if __has_include(<hip/amd_detail/amd_gfx1250_TDM.h>)
#define TDM_SIX_ARGS 1
#else
#define TDM_SIX_ARGS 0
#endif

__device__ __forceinline__ v8f wmma_bf16(v16bf a, v16bf b, v8f c) {
    // D = A(16x32 bf16) * B(32x16 bf16) + C(16x16 f32), wave32
    return __builtin_amdgcn_wmma_f32_16x16x32_bf16(false, a, false, b,
                                                   (short)0, c, false, false);
}

__device__ __forceinline__ v16bf join8(v8bf lo, v8bf hi) {
    return __builtin_shufflevector(lo, hi, 0, 1, 2, 3, 4, 5, 6, 7,
                                          8, 9, 10, 11, 12, 13, 14, 15);
}

__device__ __forceinline__ v8f vzero8() {
    v8f z = {0.f, 0.f, 0.f, 0.f, 0.f, 0.f, 0.f, 0.f};
    return z;
}

#if USE_TDM
typedef unsigned int u32x4 __attribute__((ext_vector_type(4)));
typedef int          i32x4 __attribute__((ext_vector_type(4)));
typedef int          i32x8 __attribute__((ext_vector_type(8)));

// Issue a TDM 2D tile load: tile (tileK x tileN) of a (tdim0 x tdim1) bf16
// tensor with row stride ldK elements, packed contiguously into LDS.
// D# bitfields per CDNA5 ISA.
__device__ __forceinline__ void tdm_load_tile(const bf16_t* gsrc,
                                              unsigned lds_byte,
                                              int tileK, int tileN,
                                              int tdim0, int tdim1, int ldK) {
    unsigned long long ga = (unsigned long long)(uintptr_t)gsrc;
    u32x4 g0;
    g0[0] = 1u;                                       // count=1 (valid user D#)
    g0[1] = lds_byte;                                 // lds_addr (bytes)
    g0[2] = (unsigned)(ga & 0xffffffffu);             // global_addr[31:0]
    g0[3] = (unsigned)((ga >> 32) & 0x01ffffffu)      // global_addr[56:32]
          | (2u << 30);                               // type = 2 ("image")
    i32x8 g1;
    g1[0] = (int)(1u << 16);                          // data_size=1 -> 2 bytes
    g1[1] = (int)(((unsigned)tdim0 & 0xffffu) << 16); // tensor_dim0[15:0]
    g1[2] = (int)((((unsigned)tdim0 >> 16) & 0xffffu)
          | (((unsigned)tdim1 & 0xffffu) << 16));     // td0[31:16] | td1[15:0]
    g1[3] = (int)((((unsigned)tdim1 >> 16) & 0xffffu)
          | (((unsigned)tileK & 0xffffu) << 16));     // td1[31:16] | tile_dim0
    g1[4] = (int)((unsigned)tileN & 0xffffu);         // tile_dim1 (tile_dim2=0)
    g1[5] = (int)(unsigned)ldK;                       // tensor_dim0_stride[31:0]
    g1[6] = 0;                                        // stride0[47:32]|stride1[15:0]
    g1[7] = 0;                                        // tensor_dim1_stride[47:16]
    i32x4 z4 = {0, 0, 0, 0};
#if TDM_SIX_ARGS
    i32x8 z8 = {0, 0, 0, 0, 0, 0, 0, 0};
    __builtin_amdgcn_tensor_load_to_lds(g0, g1, z4, z4, z8, 0);
#else
    __builtin_amdgcn_tensor_load_to_lds(g0, g1, z4, z4, 0);
#endif
}

__device__ __forceinline__ void tdm_wait0() {
#if __has_builtin(__builtin_amdgcn_s_wait_tensorcnt)
    __builtin_amdgcn_s_wait_tensorcnt(0);
#else
    asm volatile("s_wait_tensorcnt 0x0" ::: "memory");
#endif
}
#endif  // USE_TDM

// ---------------------------------------------------------------------------
// Elementwise converts
// ---------------------------------------------------------------------------
__global__ void k_f32_to_bf16(const float* __restrict__ in,
                              bf16_t* __restrict__ out, int n) {
    int i = blockIdx.x * blockDim.x + threadIdx.x;
    if (i < n) out[i] = (bf16_t)in[i];
}

// out[c*rows + r] = (bf16) in[r*cols + c]   (weight transpose: Wt[n][k])
__global__ void k_transpose_to_bf16(const float* __restrict__ in,
                                    bf16_t* __restrict__ out,
                                    int rows, int cols) {
    int i = blockIdx.x * blockDim.x + threadIdx.x;
    if (i >= rows * cols) return;
    int r = i / cols, c = i - r * cols;
    out[(size_t)c * rows + r] = (bf16_t)in[i];
}

// ---------------------------------------------------------------------------
// WMMA GEMM: C[M x N] = A[M x K](bf16, row-major) * Bt[N x K](bf16)^T + bias
// Block = 256 threads (8 waves).  Block tile = 128 rows x 128 cols; each wave
// owns 16 rows x 128 cols (acc = 64 VGPRs/lane -> no scratch spills).  The
// 32(K) x 128(N) B tile is double-buffered in LDS, staged by the Tensor Data
// Mover (TENSORcnt) when available, else by a cooperative copy, one k-step
// ahead of the WMMA pipeline.
// mode: 0 = f32 row-major store, 2 = bf16 transposed store (C^T, N x M).
// ---------------------------------------------------------------------------
#define BN 128
#define BM 128
#define BK 32

__global__ void __launch_bounds__(256)
k_gemm_wmma(const bf16_t* __restrict__ A,
            const bf16_t* __restrict__ Bt,
            const float*  __restrict__ bias,
            float*  __restrict__ Cf,
            bf16_t* __restrict__ Cb,
            int M, int N, int K, int mode) {
    __shared__ __align__(32) bf16_t Bs[2 * BN * BK];   // 16 KB, double buffered

    const int tid  = threadIdx.x;
    const int wave = tid >> 5;            // 0..7, owns 16 rows
    const int lane = tid & 31;
    const int nlo  = lane & 15;
    const int hh   = lane >> 4;
    const int r0w  = blockIdx.y * BM + wave * 16;
    const int c0   = blockIdx.x * BN;

    v8f acc[8];
#pragma unroll
    for (int j = 0; j < 8; ++j) acc[j] = vzero8();

    const bf16_t* arow = A + (size_t)(r0w + nlo) * K;

    const int nk = K / BK;

    // ---- stage k-step into buffer `buf` -----------------------------------
    auto stage = [&](int buf, int kc) {
#if USE_TDM
        if (tid < 32) {   // one wave issues the TDM descriptor for the block
            tdm_load_tile(Bt + (size_t)c0 * K + kc,
                          (unsigned)(uintptr_t)&Bs[buf * (BN * BK)],
                          BK, BN, K, N, K);
        }
#else
        // cooperative fallback: 256 threads copy 128 rows x 32 K elements
        const int rrow = tid >> 1, half = tid & 1;
        const bf16_t* src = Bt + (size_t)(c0 + rrow) * K + kc + half * 16;
        bf16_t*       dst = &Bs[buf * (BN * BK) + rrow * BK + half * 16];
        *(v8bf*)(dst)     = *(const v8bf*)(src);
        *(v8bf*)(dst + 8) = *(const v8bf*)(src + 8);
#endif
    };

    stage(0, 0);                               // prologue: fill buffer 0

    for (int it = 0; it < nk; ++it) {
        const int kc  = it * BK;
        const int cur = it & 1;

#if USE_TDM
        if (tid < 32) tdm_wait0();             // my buffer-`cur` DMA is done
#endif
        __syncthreads();                       // everyone's stage visible; all
                                               // reads of buf cur^1 finished
        if (it + 1 < nk) stage(cur ^ 1, kc + BK);

        // A fragment for this wave's 16 rows (direct from L2)
        v16bf a = join8(*(const v8bf*)(arow + kc + hh * 16),
                        *(const v8bf*)(arow + kc + hh * 16 + 8));
        if (it + 1 < nk)                       // stream next A through L2
            __builtin_prefetch(arow + kc + BK, 0, 1);

        const bf16_t* bbase = &Bs[cur * (BN * BK)];
#pragma unroll
        for (int j = 0; j < 8; ++j) {
            const bf16_t* bp = bbase + (j * 16 + nlo) * BK + hh * 16;
            v16bf b = join8(*(const v8bf*)bp, *(const v8bf*)(bp + 8));
            acc[j] = wmma_bf16(a, b, acc[j]);
        }
    }

#pragma unroll
    for (int j = 0; j < 8; ++j) {
        const int   col = c0 + j * 16 + nlo;
        const float bv  = bias[col];
#pragma unroll
        for (int v = 0; v < 8; ++v) {
            const int row = r0w + v + 8 * hh;         // C layout: M = v + 8*half
            float val = acc[j][v] + bv;
            if (mode == 0)
                Cf[(size_t)row * N + col] = val;
            else
                Cb[(size_t)col * M + row] = (bf16_t)val;  // transposed bf16
        }
    }
}

// ---------------------------------------------------------------------------
// RMSNorm over rows of D=1536, writes bf16:  out = in * rsqrt(mean(in^2)+eps)*g
// ---------------------------------------------------------------------------
__global__ void k_rmsnorm_bf16(const float* __restrict__ in,
                               const float* __restrict__ g,
                               bf16_t* __restrict__ out, int D) {
    __shared__ float red[256];
    const int row = blockIdx.x;
    const int tid = threadIdx.x;
    const float* r = in + (size_t)row * D;
    float ss = 0.f;
    for (int c = tid; c < D; c += 256) {
        float v = r[c];
        ss += v * v;
    }
    red[tid] = ss;
    __syncthreads();
    for (int s = 128; s > 0; s >>= 1) {
        if (tid < s) red[tid] += red[tid + s];
        __syncthreads();
    }
    float scale = rsqrtf(red[0] / (float)D + EPSV);
    for (int c = tid; c < D; c += 256)
        out[(size_t)row * D + c] = (bf16_t)(r[c] * scale * g[c]);
}

// ---------------------------------------------------------------------------
// Flash attention with the reference's block-causal/window mask.
// One wave32 per (head, 16-query tile); 32 keys per iteration.
//   qb, kb : (S x 1536) bf16 row-major (post RMSNorm)
//   vt     : (1536 x S) bf16  -- V transposed, so P*V B-frags are contiguous
//   ob     : (S x 1536) bf16 output (padded rows zeroed)
// ---------------------------------------------------------------------------
__global__ void k_attn(const bf16_t* __restrict__ qb,
                       const bf16_t* __restrict__ kb,
                       const bf16_t* __restrict__ vt,
                       const int* __restrict__ seq,
                       const int* __restrict__ fr,
                       const int* __restrict__ nz,
                       const int* __restrict__ wsz,
                       bf16_t* __restrict__ ob, int S) {
    __shared__ __align__(32) bf16_t pshm[16 * 32];   // P tile, row-major 16x32
    const int lane = threadIdx.x & 31;
    const int nlo  = lane & 15;
    const int hh   = lane >> 4;
    const int q0   = blockIdx.x * 16;
    const int head = blockIdx.y;
    const int W    = wsz[0];

    // Q fragments: 4 K-chunks of 32 along DHEAD=128, kept in registers
    v16bf qf[4];
    const bf16_t* qrow = qb + (size_t)(q0 + nlo) * DIMW + head * DHEAD;
#pragma unroll
    for (int c = 0; c < 4; ++c)
        qf[c] = join8(*(const v8bf*)(qrow + c * 32 + hh * 16),
                      *(const v8bf*)(qrow + c * 32 + hh * 16 + 8));

    // per-lane copies of query ids for the 8 rows this lane owns in C layout
    int qs[8], qfr[8], qnz[8];
#pragma unroll
    for (int v = 0; v < 8; ++v) {
        int r  = q0 + v + 8 * hh;
        qs[v]  = seq[r];
        qfr[v] = fr[r];
        qnz[v] = nz[r];
    }

    float rmax[8], rsum[8];
    v8f   acc[8];
#pragma unroll
    for (int v = 0; v < 8; ++v) { rmax[v] = -__builtin_inff(); rsum[v] = 0.f; }
#pragma unroll
    for (int o = 0; o < 8; ++o) acc[o] = vzero8();

    const float scl = 0.08838834764831845f;   // 1/sqrt(128)

    for (int kb0 = 0; kb0 < S; kb0 += 32) {
        if (kb0 + 32 < S)   // stream next key block through L2
            __builtin_prefetch(kb + (size_t)(kb0 + 32 + nlo) * DIMW
                                  + head * DHEAD, 0, 1);
        v8f st[2];
#pragma unroll
        for (int j = 0; j < 2; ++j) {
            const int kidx = kb0 + j * 16 + nlo;   // this lane's key (B col)
            const bf16_t* krow =
                kb + (size_t)kidx * DIMW + head * DHEAD + hh * 16;
            v8f sc = vzero8();
#pragma unroll
            for (int c = 0; c < 4; ++c) {
                v16bf b = *(const v16bf*)(krow + c * 32);
                sc = wmma_bf16(qf[c], b, sc);
            }
            const int sk = seq[kidx], fk = fr[kidx], nk = nz[kidx];
#pragma unroll
            for (int v = 0; v < 8; ++v) {
                bool seq_ok = (qs[v] == sk) && (qs[v] >= 0) && (sk >= 0);
                bool c2c = (qnz[v] == 1) && (nk == 1) && (fk <= qfr[v]);
                bool n2c = (qnz[v] == 0) && (nk == 1) && (fk <  qfr[v]);
                bool n2n = (qnz[v] == 0) && (nk == 0) && (fk == qfr[v]);
                int  d   = qfr[v] - fk;
                bool win = ((d < 0) ? -d : d) <= W;
                bool ok  = (c2c || n2c || n2n) && seq_ok && win;
                st[j][v] = ok ? sc[v] * scl : -1.0e9f;
            }
        }

        // online softmax; row m = v + 8*hh lives across 16 lanes of one half
#pragma unroll
        for (int v = 0; v < 8; ++v) {
            float mx = fmaxf(st[0][v], st[1][v]);
#pragma unroll
            for (int m = 1; m <= 8; m <<= 1)
                mx = fmaxf(mx, __shfl_xor(mx, m, 32));
            float nm    = fmaxf(rmax[v], mx);
            float alpha = __expf(rmax[v] - nm);
            rmax[v]     = nm;
            float p0 = __expf(st[0][v] - nm);
            float p1 = __expf(st[1][v] - nm);
            float ls = p0 + p1;
#pragma unroll
            for (int m = 1; m <= 8; m <<= 1)
                ls += __shfl_xor(ls, m, 32);
            rsum[v] = rsum[v] * alpha + ls;
#pragma unroll
            for (int o = 0; o < 8; ++o) acc[o][v] *= alpha;
            pshm[(v + 8 * hh) * 32 + nlo]      = (bf16_t)p0;
            pshm[(v + 8 * hh) * 32 + 16 + nlo] = (bf16_t)p1;
        }

        // same-wave LDS ordering: make P stores visible to the A-frag loads
        asm volatile("s_wait_dscnt 0" ::: "memory");

        v16bf pa = join8(*(const v8bf*)(pshm + nlo * 32 + hh * 16),
                         *(const v8bf*)(pshm + nlo * 32 + hh * 16 + 8));
#pragma unroll
        for (int o = 0; o < 8; ++o) {
            // B frag from V^T: col = output dim o*16+nlo, 16 consecutive keys
            const bf16_t* vrow =
                vt + (size_t)(head * DHEAD + o * 16 + nlo) * S + kb0 + hh * 16;
            v16bf b = *(const v16bf*)vrow;
            acc[o] = wmma_bf16(pa, b, acc[o]);
        }
    }

#pragma unroll
    for (int o = 0; o < 8; ++o) {
#pragma unroll
        for (int v = 0; v < 8; ++v) {
            float val = acc[o][v] / rsum[v];
            if (qs[v] < 0) val = 0.f;               // zero padded query rows
            int row = q0 + v + 8 * hh;
            ob[(size_t)row * DIMW + head * DHEAD + o * 16 + nlo] = (bf16_t)val;
        }
    }
}

// ---------------------------------------------------------------------------
// Host-side orchestration (all launches on `stream`; graph-capture safe)
// ---------------------------------------------------------------------------
extern "C" void kernel_launch(void* const* d_in, const int* in_sizes, int n_in,
                              void* d_out, int out_size, void* d_ws,
                              size_t ws_size, hipStream_t stream) {
    (void)in_sizes; (void)n_in; (void)out_size; (void)ws_size;
    const float* x   = (const float*)d_in[0];
    const float* Wq  = (const float*)d_in[1];
    const float* bq  = (const float*)d_in[2];
    const float* Wk  = (const float*)d_in[3];
    const float* bk  = (const float*)d_in[4];
    const float* Wv  = (const float*)d_in[5];
    const float* bv  = (const float*)d_in[6];
    const float* gq  = (const float*)d_in[7];
    const float* gk  = (const float*)d_in[8];
    const float* Wo  = (const float*)d_in[9];
    const float* bo  = (const float*)d_in[10];
    const int*   seq = (const int*)d_in[11];
    const int*   fr  = (const int*)d_in[12];
    const int*   nz  = (const int*)d_in[13];
    const int*   wsz = (const int*)d_in[14];
    float*       out = (float*)d_out;

    // workspace carve-out (~63 MB)
    char*  ws  = (char*)d_ws;
    size_t off = 0;
    auto carve = [&](size_t bytes) -> void* {
        void* p = ws + off;
        off += (bytes + 255) & ~(size_t)255;
        return p;
    };
    const size_t SD = (size_t)S_TOK * DIMW;
    bf16_t* xb     = (bf16_t*)carve(SD * 2);
    bf16_t* wqt    = (bf16_t*)carve((size_t)DIMW * DIMW * 2);
    bf16_t* wkt    = (bf16_t*)carve((size_t)DIMW * DIMW * 2);
    bf16_t* wvt    = (bf16_t*)carve((size_t)DIMW * DIMW * 2);
    bf16_t* wot    = (bf16_t*)carve((size_t)DIMW * DIMW * 2);
    bf16_t* qbuf   = (bf16_t*)carve(SD * 2);
    bf16_t* kbuf   = (bf16_t*)carve(SD * 2);
    float*  f32buf = (float*)carve(SD * 4);     // reused: q-pre, k-pre, then V^T + O
    bf16_t* vt     = (bf16_t*)f32buf;           // S*D bf16  (first half)
    bf16_t* obuf   = vt + SD;                   // S*D bf16  (second half)

    const int n  = (int)SD;
    const int nw = DIMW * DIMW;

    k_f32_to_bf16<<<(n + 255) / 256, 256, 0, stream>>>(x, xb, n);
    k_transpose_to_bf16<<<(nw + 255) / 256, 256, 0, stream>>>(Wq, wqt, DIMW, DIMW);
    k_transpose_to_bf16<<<(nw + 255) / 256, 256, 0, stream>>>(Wk, wkt, DIMW, DIMW);
    k_transpose_to_bf16<<<(nw + 255) / 256, 256, 0, stream>>>(Wv, wvt, DIMW, DIMW);
    k_transpose_to_bf16<<<(nw + 255) / 256, 256, 0, stream>>>(Wo, wot, DIMW, DIMW);

    dim3 ggrid(DIMW / BN, S_TOK / BM);          // (12, 24) blocks of 8 waves

    // q = rms(x@Wq + bq) * gq     (f32 staging, then normalized bf16)
    k_gemm_wmma<<<ggrid, 256, 0, stream>>>(xb, wqt, bq, f32buf, nullptr,
                                           S_TOK, DIMW, DIMW, 0);
    k_rmsnorm_bf16<<<S_TOK, 256, 0, stream>>>(f32buf, gq, qbuf, DIMW);

    // k = rms(x@Wk + bk) * gk
    k_gemm_wmma<<<ggrid, 256, 0, stream>>>(xb, wkt, bk, f32buf, nullptr,
                                           S_TOK, DIMW, DIMW, 0);
    k_rmsnorm_bf16<<<S_TOK, 256, 0, stream>>>(f32buf, gk, kbuf, DIMW);

    // v = x@Wv + bv, stored transposed (V^T, 1536 x S) in bf16
    k_gemm_wmma<<<ggrid, 256, 0, stream>>>(xb, wvt, bv, nullptr, vt,
                                           S_TOK, DIMW, DIMW, 2);

    // masked flash attention -> obuf (bf16, padded rows zeroed)
    dim3 agrid(S_TOK / 16, HEADS);
    k_attn<<<agrid, 32, 0, stream>>>(qbuf, kbuf, vt, seq, fr, nz, wsz,
                                     obuf, S_TOK);

    // y = o@Wo + bo (f32 out)
    k_gemm_wmma<<<ggrid, 256, 0, stream>>>(obuf, wot, bo, out, nullptr,
                                           S_TOK, DIMW, DIMW, 0);
}
